// MoEc_84164179132798
// MI455X (gfx1250) — matrix operand order repaired and checked by reading, not compiled, not run on hardware
//
#include <hip/hip_runtime.h>

// Problem sizes (fixed by the reference)
#define BTOK 32768   // batch
#define DIN  1024    // input dim
#define HID  512     // hidden
#define FEXP 100     // expert hidden
#define FPAD 112     // expert hidden padded to 7*16 for WMMA N tiles
#define NEXP 16      // experts
#define NSUB 7       // FPAD / 16

typedef __attribute__((ext_vector_type(16))) __bf16 v16bf;
typedef __attribute__((ext_vector_type(8)))  __bf16 v8bf;
typedef __attribute__((ext_vector_type(4)))  __bf16 v4bf;
typedef __attribute__((ext_vector_type(8)))  float  v8f;
typedef __attribute__((ext_vector_type(4)))  float  v4f;
typedef __attribute__((ext_vector_type(4)))  int    v4i;

// ---- CDNA5 async global->LDS copy (ASYNCcnt) with compile-safe fallback ----
#if defined(__has_builtin)
#  if __has_builtin(__builtin_amdgcn_global_load_async_to_lds_b128) && \
      __has_builtin(__builtin_amdgcn_s_wait_asynccnt)
#    define USE_ASYNC_LDS 1
#  endif
#endif
#ifndef USE_ASYNC_LDS
#  define USE_ASYNC_LDS 0
#endif

#define AS1 __attribute__((address_space(1)))
#define AS3 __attribute__((address_space(3)))

static __device__ __forceinline__ void async_copy16(const void* g, void* l) {
#if USE_ASYNC_LDS
    // signature (probe-confirmed): (v4i AS1* src, v4i AS3* dst, imm off, imm cpol)
    __builtin_amdgcn_global_load_async_to_lds_b128((AS1 v4i*)(void*)g,
                                                   (AS3 v4i*)l, 0, 0);
#else
    *(v8bf*)l = *(const v8bf*)g;
#endif
}
static __device__ __forceinline__ void async_wait() {
#if USE_ASYNC_LDS
    __builtin_amdgcn_s_wait_asynccnt(0);
#endif
}

// Build a 16-wide bf16 fragment from two contiguous 16-byte chunks.
static __device__ __forceinline__ v16bf frag16(const __bf16* lo, const __bf16* hi) {
    v8bf a = *(const v8bf*)lo;
    v8bf b = *(const v8bf*)hi;
    return __builtin_shufflevector(a, b, 0,1,2,3,4,5,6,7,8,9,10,11,12,13,14,15);
}

// ---------------------------------------------------------------------------
// Setup kernels: zero accumulators/counters, convert weights fp32 -> bf16
// ---------------------------------------------------------------------------
__global__ void k_zero(float* __restrict__ out, int* __restrict__ cnt) {
    int i = blockIdx.x * blockDim.x + threadIdx.x;
    if (i < BTOK) out[i] = 0.0f;
    if (blockIdx.x == 0 && threadIdx.x < NEXP) cnt[threadIdx.x] = 0;
}

__global__ void k_cvt_wsh(const float* __restrict__ W, __bf16* __restrict__ o) {
    int i = blockIdx.x * blockDim.x + threadIdx.x;
    if (i < DIN * HID) o[i] = (__bf16)W[i];
}

// W1 [E,H,F] fp32 -> [E,H,FPAD] bf16, zero padded (padded cols killed by W2=0)
__global__ void k_build_w1p(const float* __restrict__ W1, __bf16* __restrict__ o) {
    int i = blockIdx.x * blockDim.x + threadIdx.x;
    if (i >= NEXP * HID * FPAD) return;
    int f  = i % FPAD;
    int eh = i / FPAD;                       // e*HID + h
    float v = (f < FEXP) ? W1[(size_t)eh * FEXP + f] : 0.0f;
    o[i] = (__bf16)v;
}

// Per-token routes -> per-expert token lists (counts bounded by BTOK per expert)
__global__ void k_build_routes(const int* __restrict__ idx, const int* __restrict__ snd,
                               int* __restrict__ cnt, int* __restrict__ list) {
    int b = blockIdx.x * blockDim.x + threadIdx.x;
    if (b >= BTOK) return;
    int i = idx[b];
    #pragma unroll
    for (int j = 0; j < 2; ++j) {
        int e = snd[i * 2 + j];
        int p = atomicAdd(&cnt[e], 1);
        list[(size_t)e * BTOK + p] = b;
    }
}

// ---------------------------------------------------------------------------
// Stage 1: h = relu(x @ W_shared + b_shared), bf16 WMMA, h stored as bf16.
// Block = 128 threads (4 waves). Block tile 64(M) x 64(N); wave tile 16x64.
// Double-buffered LDS; W_shared tile uses async global->LDS (no conversion).
// ---------------------------------------------------------------------------
static __device__ __forceinline__ void s1_fill_A(const float* __restrict__ x,
                                                 __bf16* bufA, int m0, int k0, int tid) {
    #pragma unroll
    for (int i = 0; i < 4; ++i) {
        int lin = tid + i * 128;          // 0..511 over 64 rows x 8 float4
        int r   = lin >> 3;
        int c4  = lin & 7;
        v4f v = *(const v4f*)(x + (size_t)(m0 + r) * DIN + k0 + c4 * 4);
        v4bf p;
        p[0] = (__bf16)v[0]; p[1] = (__bf16)v[1];
        p[2] = (__bf16)v[2]; p[3] = (__bf16)v[3];
        *(v4bf*)(&bufA[r * 40 + c4 * 4]) = p;
    }
}
static __device__ __forceinline__ void s1_fill_B(const __bf16* __restrict__ Wsh,
                                                 __bf16* bufB, int n0, int k0, int tid) {
    #pragma unroll
    for (int i = 0; i < 2; ++i) {
        int lin = tid + i * 128;          // 0..255 over 32 rows x 8 chunks
        int r   = lin >> 3;
        int c8  = lin & 7;
        async_copy16(Wsh + (size_t)(k0 + r) * HID + n0 + c8 * 8,
                     &bufB[r * 72 + c8 * 8]);
    }
}

__global__ void __launch_bounds__(128)
k_shared_gemm(const float* __restrict__ x, const __bf16* __restrict__ Wsh,
              const float* __restrict__ bsh, __bf16* __restrict__ hout) {
    __shared__ __bf16 lA[2][64 * 40];   // 64 rows x 32 K, padded to 40
    __shared__ __bf16 lB[2][32 * 72];   // 32 K  x 64 N, padded to 72

    const int tid  = threadIdx.x;
    const int lane = tid & 31;
    const int wv   = tid >> 5;
    const int m0   = blockIdx.x * 64;
    const int n0   = blockIdx.y * 64;
    const int g    = lane >> 4;               // lane group (A K-offset 8*g)
    const int arow = wv * 16 + (lane & 15);   // A row in LDS tile for this lane

    v8f acc[4] = {};

    // prologue: fill buffer 0 with the k0=0 tiles
    s1_fill_A(x, lA[0], m0, 0, tid);
    s1_fill_B(Wsh, lB[0], n0, 0, tid);
    async_wait();
    __syncthreads();

    for (int k0 = 0; k0 < DIN; k0 += 32) {
        const int cur = (k0 >> 5) & 1;
        // issue next-tile fills first so global latency overlaps the WMMAs
        if (k0 + 32 < DIN) {
            s1_fill_A(x, lA[1 - cur], m0, k0 + 32, tid);
            s1_fill_B(Wsh, lB[1 - cur], n0, k0 + 32, tid);
            if (k0 + 64 < DIN)   // speculative prefetch of the k+64 x tile
                __builtin_prefetch(x + (size_t)(m0 + (tid >> 1)) * DIN + (k0 + 64), 0, 1);
        }

        // compute from current buffer: A frag + 4 B frags, then WMMA chain
        v16bf af = frag16(&lA[cur][arow * 40 + 8 * g],
                          &lA[cur][arow * 40 + 8 * g + 16]);
        v16bf bfrag[4];
        #pragma unroll
        for (int nt = 0; nt < 4; ++nt) {
            #pragma unroll
            for (int s = 0; s < 16; ++s)
                bfrag[nt][s] = lB[cur][(8 * g + (s & 7) + ((s >> 3) << 4)) * 72 + nt * 16 + (lane & 15)];
        }
        #pragma unroll
        for (int nt = 0; nt < 4; ++nt)
            acc[nt] = __builtin_amdgcn_wmma_f32_16x16x32_bf16(
                false, af, false, bfrag[nt], (short)0, acc[nt], false, false);

        async_wait();        // next-buffer async fills complete
        __syncthreads();     // + ds_stores visible to all waves
    }

    // ---- epilogue: +bias, ReLU, convert to bf16, store h ----
    const int col0 = n0 + (lane & 15);
    #pragma unroll
    for (int nt = 0; nt < 4; ++nt) {
        int   n    = col0 + nt * 16;
        float bias = bsh[n];
        #pragma unroll
        for (int r = 0; r < 8; ++r) {
            int   tok = m0 + wv * 16 + r + 8 * g;   // C layout: M = r + 8*(lane/16)
            float v   = acc[nt][r] + bias;
            v = v > 0.0f ? v : 0.0f;
            hout[(size_t)tok * HID + n] = (__bf16)v;
        }
    }
}

// ---------------------------------------------------------------------------
// Stage 2 (routed): for each (expert e, tile of 16 routed tokens):
//   a = relu(h_tile @ W1[e] + b1[e]);  o = a . W2[e] + b2[e];  out[tok] += 0.5*o
// One wave per block; 7 WMMA accumulators cover F=112; K = 512 in steps of 32.
// Gathered h rows go through async global->LDS; W1p streams straight to VGPRs.
// ---------------------------------------------------------------------------
static __device__ __forceinline__ void s2_fill_A(const __bf16* __restrict__ hbf,
                                                 const int* __restrict__ list,
                                                 size_t lbase, int nvalid,
                                                 __bf16* bufA, int k0, int lane) {
    #pragma unroll
    for (int i = 0; i < 2; ++i) {
        int lin = lane + i * 32;             // 0..63 over 16 rows x 4 chunks
        int r   = lin >> 2;
        int c8  = lin & 3;
        int rr  = r < nvalid ? r : 0;
        int tok = list[lbase + rr];
        async_copy16(hbf + (size_t)tok * HID + k0 + c8 * 8, &bufA[r * 40 + c8 * 8]);
    }
}

__global__ void __launch_bounds__(32)
k_expert_gemm(const __bf16* __restrict__ hbf, const __bf16* __restrict__ W1p,
              const float* __restrict__ b1, const float* __restrict__ W2,
              const float* __restrict__ b2, const int* __restrict__ cnt,
              const int* __restrict__ list, float* __restrict__ out) {
    __shared__ __bf16 lA[2][16 * 40];   // gathered 16-token x 32-K bf16 tile

    const int e    = blockIdx.y;
    const int tile = blockIdx.x;
    const int c    = cnt[e];                     // uniform across the wave
    if (tile * 16 >= c) return;                  // early exit for unused tiles
    const int nvalid = min(16, c - tile * 16);

    const int lane = threadIdx.x;
    const int g    = lane >> 4;
    const int n16  = lane & 15;
    const size_t lbase = (size_t)e * BTOK + tile * 16;

    v8f acc[NSUB] = {};
    const __bf16* wb = W1p + (size_t)e * HID * FPAD;

    s2_fill_A(hbf, list, lbase, nvalid, lA[0], 0, lane);
    async_wait();
    __syncthreads();

    for (int k0 = 0; k0 < HID; k0 += 32) {
        const int cur = (k0 >> 5) & 1;
        if (k0 + 32 < HID)
            s2_fill_A(hbf, list, lbase, nvalid, lA[1 - cur], k0 + 32, lane);

        v16bf af = frag16(&lA[cur][n16 * 40 + 8 * g],
                          &lA[cur][n16 * 40 + 8 * g + 16]);
        v16bf bfrag[NSUB];
        #pragma unroll
        for (int nt = 0; nt < NSUB; ++nt) {
            #pragma unroll
            for (int s = 0; s < 16; ++s)
                bfrag[nt][s] = wb[(size_t)(k0 + 8 * g + (s & 7) + ((s >> 3) << 4)) * FPAD + nt * 16 + n16];
        }
        #pragma unroll
        for (int nt = 0; nt < NSUB; ++nt)
            acc[nt] = __builtin_amdgcn_wmma_f32_16x16x32_bf16(
                false, af, false, bfrag[nt], (short)0, acc[nt], false, false);

        async_wait();
        __syncthreads();
    }

    // ---- fused epilogue: +b1, ReLU, dot W2, cross-lane reduce, route-avg ----
    float w2v[NSUB], b1v[NSUB];
    #pragma unroll
    for (int nt = 0; nt < NSUB; ++nt) {
        int  f  = nt * 16 + n16;
        bool ok = f < FEXP;
        w2v[nt] = ok ? W2[e * FEXP + f] : 0.0f;
        b1v[nt] = ok ? b1[e * FEXP + f] : 0.0f;
    }
    const float bias2 = b2[e];
    #pragma unroll
    for (int r = 0; r < 8; ++r) {
        float s = 0.0f;
        #pragma unroll
        for (int nt = 0; nt < NSUB; ++nt) {
            float a = acc[nt][r] + b1v[nt];
            a = a > 0.0f ? a : 0.0f;
            s += a * w2v[nt];
        }
        // reduce over the 16 lanes holding the F dimension (fixed tree: det.)
        s += __shfl_xor(s, 1, 32);
        s += __shfl_xor(s, 2, 32);
        s += __shfl_xor(s, 4, 32);
        s += __shfl_xor(s, 8, 32);
        int mr = r + 8 * g;                      // row within the 16-token tile
        if (n16 == 0 && mr < nvalid) {
            int tok = list[lbase + mr];
            atomicAdd(&out[tok], 0.5f * (s + bias2));   // 2 adds/token, order-inv.
        }
    }
}

// ---------------------------------------------------------------------------
extern "C" void kernel_launch(void* const* d_in, const int* in_sizes, int n_in,
                              void* d_out, int out_size, void* d_ws, size_t ws_size,
                              hipStream_t stream) {
    (void)in_sizes; (void)n_in; (void)out_size; (void)ws_size;
    const float* x    = (const float*)d_in[0];
    const int*   idx  = (const int*)  d_in[1];
    const float* Wsh  = (const float*)d_in[2];
    const float* bsh  = (const float*)d_in[3];
    const float* W1   = (const float*)d_in[4];
    const float* b1   = (const float*)d_in[5];
    const float* W2   = (const float*)d_in[6];
    const float* b2   = (const float*)d_in[7];
    const int*   snd  = (const int*)  d_in[8];
    float*       out  = (float*)d_out;

    // workspace layout (~38.4 MB total)
    char*   ws    = (char*)d_ws;
    __bf16* wshb  = (__bf16*)(ws);                                   //  1.00 MB
    __bf16* w1p   = (__bf16*)(ws + (size_t)DIN * HID * 2);           //  1.79 MB
    __bf16* hbf   = (__bf16*)((char*)w1p + (size_t)NEXP * HID * FPAD * 2); // 33.55 MB
    int*    list  = (int*)  ((char*)hbf + (size_t)BTOK * HID * 2);   //  2.00 MB
    int*    cnt   = (int*)  ((char*)list + (size_t)NEXP * BTOK * 4); //  64 B

    k_zero        <<<(BTOK + 255) / 256, 256, 0, stream>>>(out, cnt);
    k_cvt_wsh     <<<(DIN * HID + 255) / 256, 256, 0, stream>>>(Wsh, wshb);
    k_build_w1p   <<<(NEXP * HID * FPAD + 255) / 256, 256, 0, stream>>>(W1, w1p);
    k_build_routes<<<(BTOK + 255) / 256, 256, 0, stream>>>(idx, snd, cnt, list);

    k_shared_gemm<<<dim3(BTOK / 64, HID / 64), 128, 0, stream>>>(x, wshb, bsh, hbf);
    k_expert_gemm<<<dim3(BTOK / 16, NEXP), 32, 0, stream>>>(hbf, w1p, b1, W2, b2,
                                                            cnt, list, out);
}
